// SpGraphAttentionLayer_23613730193909
// MI455X (gfx1250) — matrix operand order repaired and checked by reading, not compiled, not run on hardware
//
#include <hip/hip_runtime.h>
#include <math.h>

// ---------------------------------------------------------------------------
// Fused sparse graph-attention (GAT) layer for MI455X / gfx1250.
//   edge_h = [x_src | x_dst | e_embed]      (16-edge tiles, double-buffered LDS)
//   edge_m = edge_h @ a.T                   (v_wmma_f32_16x16x32_bf16, f32 acc)
//   logit  = edge_h @ (a.T @ a2)            (folded: exact, f32, fused in gather)
//   ee     = exp(-leakyrelu(logit))
//   scatter-add ee and ee*edge_m by src     (L2-resident f32 atomics)
//   out    = elu(h / rowsum)
// Software pipeline: next tile's global loads issue BEFORE current tile's WMMA;
// convert+LDS-store after it. One barrier per tile.
// ---------------------------------------------------------------------------

typedef __attribute__((ext_vector_type(16))) __bf16 v16bf;
typedef __attribute__((ext_vector_type(8)))  __bf16 v8bf;
typedef __attribute__((ext_vector_type(8)))  float  v8f;

#define F_IN       128
#define R_DIM      128
#define F_OUT      128
#define K_TOT      384          // 2*F_IN + R
#define K_STEPS    12           // 384 / 32
#define LDS_STRIDE 392          // 384 + 8 pad (bf16 units) -> conflict-free ds reads
#define LRELU_A    0.2f

// ---------------------------------------------------------------------------
__global__ void k_zero(float* __restrict__ p, long long n) {
  long long i = (long long)blockIdx.x * blockDim.x + threadIdx.x;
  long long stride = (long long)gridDim.x * blockDim.x;
  for (; i < n; i += stride) p[i] = 0.0f;
}

// Convert weight matrix a [F_OUT][K_TOT] f32 -> bf16 (row-major, same layout)
__global__ void k_prep_a(const float* __restrict__ a,
                         unsigned short* __restrict__ abf_u, int n) {
  int i = blockIdx.x * blockDim.x + threadIdx.x;
  if (i < n) ((__bf16*)abf_u)[i] = (__bf16)a[i];
}

// Fold the logit projection: w2[k] = sum_n a2[n] * a[n][k]   (exact, linear)
__global__ void k_prep_w2(const float* __restrict__ a,
                          const float* __restrict__ a2,
                          float* __restrict__ w2) {
  int k = blockIdx.x * blockDim.x + threadIdx.x;
  if (k < K_TOT) {
    float s = 0.0f;
    for (int n = 0; n < F_OUT; ++n) s = fmaf(a2[n], a[n * K_TOT + k], s);
    w2[k] = s;
  }
}

// ---------------------------------------------------------------------------
__device__ __forceinline__ float dot8(float4 a0, float4 a1, float4 b0, float4 b1) {
  float d = a0.x * b0.x;
  d = fmaf(a0.y, b0.y, d); d = fmaf(a0.z, b0.z, d); d = fmaf(a0.w, b0.w, d);
  d = fmaf(a1.x, b1.x, d); d = fmaf(a1.y, b1.y, d);
  d = fmaf(a1.z, b1.z, d); d = fmaf(a1.w, b1.w, d);
  return d;
}

__device__ __forceinline__ v8bf cvt8(float4 a0, float4 a1) {
  v8bf b;
  b[0] = (__bf16)a0.x; b[1] = (__bf16)a0.y; b[2] = (__bf16)a0.z; b[3] = (__bf16)a0.w;
  b[4] = (__bf16)a1.x; b[5] = (__bf16)a1.y; b[6] = (__bf16)a1.z; b[7] = (__bf16)a1.w;
  return b;
}

// ---------------------------------------------------------------------------
__global__ void __launch_bounds__(256)
gat_main(const float* __restrict__ input,          // [N][F_IN]
         const int* __restrict__ edge,             // [2][E]
         const float* __restrict__ edge_embed,     // [E][R_DIM]
         const unsigned short* __restrict__ abf_u, // bf16 [F_OUT][K_TOT]
         const float* __restrict__ w2,             // [K_TOT] = a.T @ a2
         float* __restrict__ hout,                 // [N][F_OUT] accumulator
         float* __restrict__ rowsum,               // [N]
         long long E, int numTiles)
{
  const __bf16* __restrict__ abf = (const __bf16*)abf_u;

  __shared__ __bf16 ldsA[2][16 * LDS_STRIDE];  // double-buffered edge_h tiles
  __shared__ int    sSrc[2][16];
  __shared__ float  sTile[2][16];              // f32 logit accumulators
  __shared__ float  eeTile[2][16];             // exp(-leakyrelu(logit))

  const int t    = threadIdx.x;
  const int lane = t & 31;
  const int wave = t >> 5;          // 0..7, owns output cols [wave*16, wave*16+16)
  const int half = lane >> 4;       // 0 / 1
  const int l16  = lane & 15;
  const int n0   = wave * 16;

  // ---- B fragments (bf16 weights) register-resident, reused for all tiles.
  // B[k][n] = a[n][k]; lane holds col n = n0+l16, K = 32*ks + half*16 + {0..15}.
  v16bf bfrag[K_STEPS];
  {
    const __bf16* bp = abf + (long long)(n0 + l16) * K_TOT + half * 16;
#pragma unroll
    for (int k = 0; k < K_STEPS; ++k)
      bfrag[k] = *(const v16bf*)(bp + k * 32);
  }

  // ---- per-thread gather-task geometry + w2 chunks (loop-invariant)
  // 16 rows x 48 chunks of 8 floats = 768 tasks, 3 per thread.
  int rr[3], kk0[3];
  float4 w2lo[3], w2hi[3];
#pragma unroll
  for (int it = 0; it < 3; ++it) {
    int q   = t + it * 256;          // 0..767
    rr[it]  = q / 48;
    kk0[it] = (q - rr[it] * 48) * 8; // 0..376, step 8
    w2lo[it] = ((const float4*)(w2 + kk0[it]))[0];
    w2hi[it] = ((const float4*)(w2 + kk0[it]))[1];
  }

  if (t < 16) { sTile[0][t] = 0.0f; sTile[1][t] = 0.0f; }
  __syncthreads();

  const int stride = gridDim.x;
  int tile = blockIdx.x;
  int p = 0;

  // ---- prologue: full gather of the first tile into buffer 0
  {
    const long long e0 = (long long)tile * 16;
#pragma unroll
    for (int it = 0; it < 3; ++it) {
      long long e  = e0 + rr[it];
      long long ec = (e < E) ? e : (E - 1);
      const int k0 = kk0[it];
      const float* src;
      int idx = 0;
      if (k0 < F_IN)          { idx = edge[ec];     src = input + (long long)idx * F_IN + k0; }
      else if (k0 < 2 * F_IN) { idx = edge[E + ec]; src = input + (long long)idx * F_IN + (k0 - F_IN); }
      else                    { src = edge_embed + ec * R_DIM + (k0 - 2 * F_IN); }
      float4 f0 = ((const float4*)src)[0];
      float4 f1 = ((const float4*)src)[1];
      *(v8bf*)&ldsA[0][rr[it] * LDS_STRIDE + k0] = cvt8(f0, f1);
      atomicAdd(&sTile[0][rr[it]], dot8(f0, f1, w2lo[it], w2hi[it]));
      if (k0 == 0) sSrc[0][rr[it]] = idx;
    }
  }
  __syncthreads();

  // ---- pipelined tile loop: ONE barrier per tile
  for (; tile < numTiles; tile += stride, p ^= 1) {
    const long long e0 = (long long)tile * 16;
    const int nb = p ^ 1;
    const bool hasNext = (tile + stride) < numTiles;

    // (a) attention weights for the current tile (reads + re-zeroes sTile[p])
    if (t < 16) {
      float s = sTile[p][t];
      sTile[p][t] = 0.0f;
      long long e = e0 + t;
      float pw = -((s > 0.0f) ? s : LRELU_A * s);
      float ee = (e < E) ? __expf(pw) : 0.0f;
      eeTile[p][t] = ee;
      if (e < E) atomicAdd(&rowsum[sSrc[p][t]], ee);   // global_atomic_add_f32
    }

    // register-stage the 8 src indices this lane's scatter needs (sSrc[p] is
    // overwritten by the NEXT iteration's store phase, after the barrier)
    int4 si0 = *(const int4*)&sSrc[p][half * 8];
    int4 si1 = *(const int4*)&sSrc[p][half * 8 + 4];

    // (b) issue next tile's global loads now -> overlap with WMMA below
    float4 f0[3], f1[3];
    int idxr[3];
    if (hasNext) {
      const long long e0n = (long long)(tile + stride) * 16;
#pragma unroll
      for (int it = 0; it < 3; ++it) {
        long long e  = e0n + rr[it];
        long long ec = (e < E) ? e : (E - 1);
        const int k0 = kk0[it];
        const float* src;
        int idx = 0;
        if (k0 < F_IN)          { idx = edge[ec];     src = input + (long long)idx * F_IN + k0; }
        else if (k0 < 2 * F_IN) { idx = edge[E + ec]; src = input + (long long)idx * F_IN + (k0 - F_IN); }
        else                    { src = edge_embed + ec * R_DIM + (k0 - 2 * F_IN); }
        f0[it] = ((const float4*)src)[0];
        f1[it] = ((const float4*)src)[1];
        idxr[it] = idx;
      }
    }

    // (c) edge_m tile = A(16x384) x B(384x16), f32 accumulate, from ldsA[p]
    // A layout: lane holds row m = l16; K = 32*ks + half*16 + {0..15}.
    v8f acc = {0.f, 0.f, 0.f, 0.f, 0.f, 0.f, 0.f, 0.f};
#pragma unroll
    for (int k = 0; k < K_STEPS; ++k) {
      const __bf16* ap = &ldsA[p][l16 * LDS_STRIDE + k * 32 + half * 16];
      v8bf lo = *(const v8bf*)ap;
      v8bf hi = *(const v8bf*)(ap + 8);
      v16bf afrag = __builtin_shufflevector(lo, hi,
          0, 1, 2, 3, 4, 5, 6, 7, 8, 9, 10, 11, 12, 13, 14, 15);
      acc = __builtin_amdgcn_wmma_f32_16x16x32_bf16(
          /*neg_a=*/false, afrag, /*neg_b=*/false, bfrag[k],
          /*c_mod=*/(short)0, acc, /*reuse_a=*/false, /*reuse_b=*/false);
    }

    // (d) stage next tile into the other buffer + fused f32 logit partials
    if (hasNext) {
#pragma unroll
      for (int it = 0; it < 3; ++it) {
        const int r = rr[it], k0 = kk0[it];
        *(v8bf*)&ldsA[nb][r * LDS_STRIDE + k0] = cvt8(f0[it], f1[it]);
        atomicAdd(&sTile[nb][r], dot8(f0[it], f1[it], w2lo[it], w2hi[it]));
        if (k0 == 0) sSrc[nb][r] = idxr[it];
      }
    }

    __syncthreads();   // eeTile[p] visible; next buffer (ldsA/sTile/sSrc) complete

    // (e) weighted message scatter: hout[src] += ee * edge_m
    // D layout: acc[r] = edge_m[m = half*8 + r][n = n0 + l16]
    const int srcs[8] = {si0.x, si0.y, si0.z, si0.w, si1.x, si1.y, si1.z, si1.w};
#pragma unroll
    for (int r = 0; r < 8; ++r) {
      const int m = half * 8 + r;
      float w = eeTile[p][m] * acc[r];
      unsigned off = (unsigned)srcs[r] * (unsigned)F_OUT + (unsigned)(n0 + l16);
      atomicAdd(&hout[off], w);   // 32-bit offset -> saddr+voffset addressing
    }
  }
}

// ---------------------------------------------------------------------------
__global__ void k_final(float* __restrict__ hout,
                        const float* __restrict__ rowsum, long long total) {
  long long i = (long long)blockIdx.x * blockDim.x + threadIdx.x;
  if (i < total) {
    float rs = rowsum[i >> 7];               // / F_OUT
    rs = (rs == 0.0f) ? 1e-12f : rs;
    float h = hout[i] / rs;
    hout[i] = (h > 0.0f) ? h : (__expf(h) - 1.0f);   // elu, alpha = 1
  }
}

// ---------------------------------------------------------------------------
extern "C" void kernel_launch(void* const* d_in, const int* in_sizes, int n_in,
                              void* d_out, int out_size, void* d_ws, size_t ws_size,
                              hipStream_t stream) {
  const float* input      = (const float*)d_in[0];
  const int*   edge       = (const int*)d_in[1];   // jax int64 demotes to int32 (x64 off)
  const float* edge_embed = (const float*)d_in[2];
  const float* a          = (const float*)d_in[3];
  const float* a2         = (const float*)d_in[4];
  float* hout = (float*)d_out;

  const long long N = (long long)in_sizes[0] / F_IN;
  const long long E = (long long)in_sizes[1] / 2;
  const int numTiles = (int)((E + 15) / 16);

  char* ws = (char*)d_ws;
  float* rowsum = (float*)ws;
  size_t off0 = ((size_t)N * sizeof(float) + 255) & ~(size_t)255;
  unsigned short* abf = (unsigned short*)(ws + off0);
  size_t off1 = off0 + (((size_t)F_OUT * K_TOT * 2 + 255) & ~(size_t)255);
  float* w2 = (float*)(ws + off1);

  // 1) zero accumulators (harness poisons buffers; we must init every call)
  const long long hN = N * F_OUT;
  k_zero<<<2048, 256, 0, stream>>>(hout, hN);
  k_zero<<<256, 256, 0, stream>>>(rowsum, N);

  // 2) weights -> bf16; fold logit projection w2 = a.T @ a2
  const int an = F_OUT * K_TOT;
  k_prep_a<<<(an + 255) / 256, 256, 0, stream>>>(a, abf, an);
  k_prep_w2<<<(K_TOT + 255) / 256, 256, 0, stream>>>(a, a2, w2);

  // 3) fused edge pass (persistent blocks; pipelined double-buffered tiles)
  int grid = numTiles < 4096 ? numTiles : 4096;
  gat_main<<<grid, 256, 0, stream>>>(input, edge, edge_embed, abf, w2,
                                     hout, rowsum, E, numTiles);

  // 4) normalize + ELU
  k_final<<<(int)((hN + 255) / 256), 256, 0, stream>>>(hout, rowsum, hN);
}
